// Block_81956565943135
// MI455X (gfx1250) — matrix operand order repaired and checked by reading, not compiled, not run on hardware
//
#include <hip/hip_runtime.h>
#include <hip/hip_bf16.h>

#define EMBED   384
#define NHEADS  6
#define HEADSZ  64
#define BATCH   4
#define SEQ     2048
#define MROWS   (BATCH * SEQ)   // 8192

typedef __attribute__((ext_vector_type(16))) _Float16 v16h;
typedef __attribute__((ext_vector_type(8)))  _Float16 v8h;
typedef __attribute__((ext_vector_type(8)))  float    v8f;
typedef int i32x4v __attribute__((vector_size(16)));

#define AS1 __attribute__((address_space(1)))
#define AS3 __attribute__((address_space(3)))

#if defined(__has_builtin)
#if __has_builtin(__builtin_amdgcn_global_load_async_to_lds_b128) && \
    __has_builtin(__builtin_amdgcn_s_wait_asynccnt)
#define USE_ASYNC_LDS 1
#endif
#endif

static __device__ __forceinline__ v16h cat8(v8h lo, v8h hi) {
  return __builtin_shufflevector(lo, hi, 0,1,2,3,4,5,6,7,8,9,10,11,12,13,14,15);
}
static __device__ __forceinline__ v8f wmma_f16(v16h a, v16h b, v8f c) {
  // D(f32 16x16) = A(16x32 f16) * B(32x16 f16) + C
  return __builtin_amdgcn_wmma_f32_16x16x32_f16(false, a, false, b, (short)0, c,
                                                false, false);
}

// ---------------------------------------------------------------------------
// f32 -> f16 conversion (weights + activations)
// ---------------------------------------------------------------------------
__global__ void cvt_f16_kernel(const float* __restrict__ src,
                               _Float16* __restrict__ dst, int n) {
  int i = blockIdx.x * blockDim.x + threadIdx.x;
  if (i < n) dst[i] = (_Float16)src[i];
}

// ---------------------------------------------------------------------------
// WMMA GEMM: C[M,N] = A[M,K](f16) * W[N,K]^T(f16) (+bias)(+relu)
// One 192-thread block owns a 16-row tile of A, staged ONCE into LDS via
// async global->LDS copy (ASYNCcnt path); each of the 6 waves computes a
// 16x64 stripe. A-fragments come from LDS (bank-conflict-free 784B row
// stride), B-fragments stream from global (weights are L2/WGP$-resident).
// Specialized for K = N = EMBED = 384.
// ---------------------------------------------------------------------------
#define LDSA_STRIDE 392   // halves per LDS row (384 data + 8 pad) = 784 B

template<bool HAS_BIAS, bool RELU, bool STORE_F16>
__global__ __launch_bounds__(192)
void gemm_wmma_kernel(const _Float16* __restrict__ A,
                      const _Float16* __restrict__ W,
                      const float* __restrict__ bias,
                      float* __restrict__ Cf,
                      _Float16* __restrict__ Ch) {
  __shared__ _Float16 As[16 * LDSA_STRIDE];   // 12,544 B
  const int tid  = threadIdx.x;
  const int lane = tid & 31;
  const int wave = tid >> 5;       // 0..5  -> 64-col group
  const int ml   = lane & 15;
  const int half = lane >> 4;      // 0: lanes 0-15, 1: lanes 16-31
  const int kbA  = half * 8;       // A fragment K sub-offset
  const int kbB  = half * 16;      // unused (B fragments use kbA)
  const int mt   = blockIdx.x;     // 16-row tile index

  // ---- stage the 16x384 A tile (contiguous 12,288 B in global) into LDS ----
  const _Float16* atile = A + (size_t)mt * 16 * EMBED;
#pragma unroll
  for (int j = 0; j < 4; ++j) {
    int c    = tid + j * 192;          // 768 x 16B chunks
    int row  = c / 48;
    int colb = (c % 48) * 16;          // byte offset within row
#ifdef USE_ASYNC_LDS
    __builtin_amdgcn_global_load_async_to_lds_b128(
        (AS1 i32x4v*)((const char*)atile + row * 768 + colb),
        (AS3 i32x4v*)((char*)As + row * (LDSA_STRIDE * 2) + colb),
        0, 0);
#else
    *(v8h*)((char*)As + row * (LDSA_STRIDE * 2) + colb) =
        *(const v8h*)((const char*)atile + row * 768 + colb);
#endif
  }
#ifdef USE_ASYNC_LDS
  __builtin_amdgcn_s_wait_asynccnt(0);
#endif
  __syncthreads();

  const _Float16* arow  = As + ml * LDSA_STRIDE;             // LDS, per-lane row
  const _Float16* wbase = W + (size_t)(wave * 64 + ml) * EMBED;

  v8f acc[4] = {v8f{}, v8f{}, v8f{}, v8f{}};
  for (int k0 = 0; k0 < EMBED; k0 += 32) {
    __builtin_prefetch(wbase + k0 + 256, 0, 0);              // global_prefetch_b8
    v16h a = cat8(*(const v8h*)(arow + k0 + kbA),            // ds_load_b128 x2
                  *(const v8h*)(arow + k0 + 16 + kbA));
#pragma unroll
    for (int s = 0; s < 4; ++s) {
      const _Float16* wrow = wbase + (size_t)(s * 16) * EMBED;
      v16h b = cat8(*(const v8h*)(wrow + k0 + kbA),
                    *(const v8h*)(wrow + k0 + 16 + kbA));
      acc[s] = wmma_f16(a, b, acc[s]);
    }
  }
#pragma unroll
  for (int s = 0; s < 4; ++s) {
    int ncol = wave * 64 + s * 16 + ml;
    float bv = HAS_BIAS ? bias[ncol] : 0.0f;
#pragma unroll
    for (int r = 0; r < 8; ++r) {
      int mrow = mt * 16 + r + half * 8;    // C-tile row mapping
      float v = acc[s][r] + bv;
      if (RELU) v = fmaxf(v, 0.0f);
      if (STORE_F16) Ch[(size_t)mrow * EMBED + ncol] = (_Float16)v;
      else           Cf[(size_t)mrow * EMBED + ncol] = v;
    }
  }
}

// ---------------------------------------------------------------------------
// q[B*T, 384] (f16, head-interleaved) -> qT[(b,h,d), t] (f16) for PV B-frags
// ---------------------------------------------------------------------------
__global__ void transpose_q_kernel(const _Float16* __restrict__ q,
                                   _Float16* __restrict__ qT) {
  int idx = blockIdx.x * blockDim.x + threadIdx.x;
  if (idx >= MROWS * EMBED) return;
  int col = idx % EMBED;
  int bt  = idx / EMBED;
  int b = bt / SEQ, t = bt % SEQ;
  int h = col / HEADSZ, d = col % HEADSZ;
  qT[(((size_t)b * NHEADS + h) * HEADSZ + d) * SEQ + t] = q[idx];
}

// ---------------------------------------------------------------------------
// Flash attention (causal, q==k==v per the reference quirk).
// One wave per (b, h, 16-query-row tile); 32-key chunks; 8 WMMA per chunk.
// Scores never touch HBM (400 MB avoided).
// ---------------------------------------------------------------------------
__global__ void attn_kernel(const _Float16* __restrict__ q,   // [B*T, 384]
                            const _Float16* __restrict__ qT,  // [(b,h,d), T]
                            _Float16* __restrict__ out) {     // [B*T, 384]
  __shared__ _Float16 plds[8][16 * 48];   // per-wave P scratch, 96B row stride
  const int lane = threadIdx.x & 31;
  const int wave = threadIdx.x >> 5;
  const int ml   = lane & 15;
  const int half = lane >> 4;
  const int kbA  = half * 8;
  const int kbB  = half * 16;

  int gw = blockIdx.x * (blockDim.x >> 5) + wave;
  const int qtPerBH = SEQ / 16;
  int qt = gw % qtPerBH;
  int bh = gw / qtPerBH;
  int h = bh % NHEADS;
  int b = bh / NHEADS;
  if (b >= BATCH) return;
  const int   m0    = qt * 16;
  const float SCALE = 0.125f;             // 1/sqrt(64)

  const _Float16* qbase = q + (size_t)b * SEQ * EMBED + (size_t)h * HEADSZ;
  const _Float16* qrow  = qbase + (size_t)(m0 + ml) * EMBED;
  // A-fragments of the query tile: K(d)=0..31 and 32..63
  v16h aq0 = cat8(*(const v8h*)(qrow + kbA),      *(const v8h*)(qrow + 16 + kbA));
  v16h aq1 = cat8(*(const v8h*)(qrow + 32 + kbA), *(const v8h*)(qrow + 48 + kbA));

  const _Float16* qTbase = qT + (((size_t)b * NHEADS + h) * HEADSZ) * SEQ;

  v8f o[4] = {v8f{}, v8f{}, v8f{}, v8f{}};
  float mrow[8], lrow[8];
#pragma unroll
  for (int r = 0; r < 8; ++r) { mrow[r] = -1e30f; lrow[r] = 0.0f; }

  _Float16* pl = &plds[wave][0];

  for (int s0 = 0; s0 <= m0 + 15; s0 += 32) {
    // speculative warm-up of next chunk's key rows (droppable prefetch)
    __builtin_prefetch(qbase + (size_t)(s0 + 32 + ml) * EMBED, 0, 0);
    // ---- scores S = q * q^T (two 16x16 tiles covering 32 keys) ----
    v8f c0 = {}, c1 = {};
    {
      const _Float16* k0r = qbase + (size_t)(s0 + ml) * EMBED;
      v16h b0 = cat8(*(const v8h*)(k0r + kbA),      *(const v8h*)(k0r + 16 + kbA));
      v16h b1 = cat8(*(const v8h*)(k0r + 32 + kbA), *(const v8h*)(k0r + 48 + kbA));
      c0 = wmma_f16(aq0, b0, c0);
      c0 = wmma_f16(aq1, b1, c0);
      const _Float16* k1r = k0r + (size_t)16 * EMBED;
      v16h b2 = cat8(*(const v8h*)(k1r + kbA),      *(const v8h*)(k1r + 16 + kbA));
      v16h b3 = cat8(*(const v8h*)(k1r + 32 + kbA), *(const v8h*)(k1r + 48 + kbA));
      c1 = wmma_f16(aq0, b2, c1);
      c1 = wmma_f16(aq1, b3, c1);
    }
    // ---- scale + causal mask + online softmax ----
    float p0[8], p1[8], cm[8], al[8], rs[8];
#pragma unroll
    for (int r = 0; r < 8; ++r) {
      int qi = m0 + r + half * 8;
      float s0v = c0[r] * SCALE; if (s0 + ml      > qi) s0v = -1e30f;
      float s1v = c1[r] * SCALE; if (s0 + 16 + ml > qi) s1v = -1e30f;
      p0[r] = s0v; p1[r] = s1v;
      cm[r] = fmaxf(s0v, s1v);
    }
#pragma unroll
    for (int off = 1; off < 16; off <<= 1)
#pragma unroll
      for (int r = 0; r < 8; ++r) cm[r] = fmaxf(cm[r], __shfl_xor(cm[r], off, 32));
#pragma unroll
    for (int r = 0; r < 8; ++r) {
      float mnew = fmaxf(mrow[r], cm[r]);
      al[r] = __expf(mrow[r] - mnew);
      mrow[r] = mnew;
      float e0 = __expf(p0[r] - mnew);
      float e1 = __expf(p1[r] - mnew);
      p0[r] = e0; p1[r] = e1;
      rs[r] = e0 + e1;
    }
#pragma unroll
    for (int off = 1; off < 16; off <<= 1)
#pragma unroll
      for (int r = 0; r < 8; ++r) rs[r] += __shfl_xor(rs[r], off, 32);
#pragma unroll
    for (int r = 0; r < 8; ++r) {
      lrow[r] = lrow[r] * al[r] + rs[r];
      o[0][r] *= al[r]; o[1][r] *= al[r]; o[2][r] *= al[r]; o[3][r] *= al[r];
    }
    // ---- re-layout P (C-layout -> A-fragment) through per-wave LDS ----
#pragma unroll
    for (int r = 0; r < 8; ++r) {
      int row = r + half * 8;
      pl[row * 48 + ml]      = (_Float16)p0[r];
      pl[row * 48 + 16 + ml] = (_Float16)p1[r];
    }
    v16h ap = cat8(*(const v8h*)(pl + ml * 48 + kbA),
                   *(const v8h*)(pl + ml * 48 + 16 + kbA));
    // ---- O += P * V (V rows are q rows s0..s0+31; contiguous via qT) ----
#pragma unroll
    for (int dt = 0; dt < 4; ++dt) {
      const _Float16* vcol = qTbase + (size_t)(dt * 16 + ml) * SEQ + s0 + kbA;
      v16h bv = cat8(*(const v8h*)(vcol), *(const v8h*)(vcol + 16));
      o[dt] = wmma_f16(ap, bv, o[dt]);
    }
  }
  // ---- normalize and store head output (concat layout) ----
  _Float16* obase = out + (size_t)b * SEQ * EMBED + (size_t)h * HEADSZ;
#pragma unroll
  for (int r = 0; r < 8; ++r) {
    float inv = 1.0f / lrow[r];
    int row = m0 + r + half * 8;
    _Float16* orow = obase + (size_t)row * EMBED;
    orow[ 0 + ml] = (_Float16)(o[0][r] * inv);
    orow[16 + ml] = (_Float16)(o[1][r] * inv);
    orow[32 + ml] = (_Float16)(o[2][r] * inv);
    orow[48 + ml] = (_Float16)(o[3][r] * inv);
  }
}

// ---------------------------------------------------------------------------
// out = LayerNorm(X + Y) * gamma + beta   (one 128-thread block per row)
// Emits f32 (residual chain) and optionally f16 (next GEMM input).
// ---------------------------------------------------------------------------
__global__ void add_ln_kernel(const float* __restrict__ X,
                              const float* __restrict__ Y,
                              const float* __restrict__ gamma,
                              const float* __restrict__ beta,
                              float* __restrict__ outF,
                              _Float16* __restrict__ outH) {
  __shared__ float sh[4];
  int row = blockIdx.x;
  const float* xr = X + (size_t)row * EMBED;
  const float* yr = Y + (size_t)row * EMBED;
  float v[3];
  float s = 0.0f;
#pragma unroll
  for (int i = 0; i < 3; ++i) {
    int c = threadIdx.x + i * 128;
    v[i] = xr[c] + yr[c];
    s += v[i];
  }
#pragma unroll
  for (int off = 16; off > 0; off >>= 1) s += __shfl_xor(s, off, 32);
  if ((threadIdx.x & 31) == 0) sh[threadIdx.x >> 5] = s;
  __syncthreads();
  float mu = (sh[0] + sh[1] + sh[2] + sh[3]) * (1.0f / EMBED);
  __syncthreads();
  float vs = 0.0f;
#pragma unroll
  for (int i = 0; i < 3; ++i) { v[i] -= mu; vs += v[i] * v[i]; }
#pragma unroll
  for (int off = 16; off > 0; off >>= 1) vs += __shfl_xor(vs, off, 32);
  if ((threadIdx.x & 31) == 0) sh[threadIdx.x >> 5] = vs;
  __syncthreads();
  float var  = (sh[0] + sh[1] + sh[2] + sh[3]) * (1.0f / EMBED);
  float rstd = rsqrtf(var + 1e-5f);
#pragma unroll
  for (int i = 0; i < 3; ++i) {
    int c = threadIdx.x + i * 128;
    float o = v[i] * rstd * gamma[c] + beta[c];
    outF[(size_t)row * EMBED + c] = o;
    if (outH) outH[(size_t)row * EMBED + c] = (_Float16)o;
  }
}

// ---------------------------------------------------------------------------
extern "C" void kernel_launch(void* const* d_in, const int* in_sizes, int n_in,
                              void* d_out, int out_size, void* d_ws, size_t ws_size,
                              hipStream_t stream) {
  (void)in_sizes; (void)n_in; (void)out_size; (void)ws_size;
  const float* x   = (const float*)d_in[0];
  const float* Wq  = (const float*)d_in[1];
  const float* Wo  = (const float*)d_in[2];
  const float* bo  = (const float*)d_in[3];
  const float* W1  = (const float*)d_in[4];
  const float* b1  = (const float*)d_in[5];
  const float* W2  = (const float*)d_in[6];
  const float* b2  = (const float*)d_in[7];
  const float* g1  = (const float*)d_in[8];
  const float* be1 = (const float*)d_in[9];
  const float* g2  = (const float*)d_in[10];
  const float* be2 = (const float*)d_in[11];

  char* w = (char*)d_ws;
  size_t off = 0;
  auto take = [&](size_t bytes) -> char* {
    char* p = w + off;
    off += (bytes + 255) & ~(size_t)255;
    return p;
  };
  const size_t H16 = (size_t)MROWS * EMBED * sizeof(_Float16);
  const size_t F32 = (size_t)MROWS * EMBED * sizeof(float);
  const size_t WB  = (size_t)EMBED * EMBED * sizeof(_Float16);

  _Float16* P0 = (_Float16*)take(H16);  // x16 -> att16
  _Float16* P1 = (_Float16*)take(H16);  // q16 -> h1(f16)
  _Float16* P2 = (_Float16*)take(H16);  // qT  -> ff1(f16)
  float*    F0 = (float*)take(F32);     // sa  -> ff2
  float*    F1 = (float*)take(F32);     // h1(f32)
  _Float16* Wq16 = (_Float16*)take(WB);
  _Float16* Wo16 = (_Float16*)take(WB);
  _Float16* W116 = (_Float16*)take(WB);
  _Float16* W216 = (_Float16*)take(WB);

  const int n_x = MROWS * EMBED;          // 3,145,728
  const int n_w = EMBED * EMBED;          // 147,456
  cvt_f16_kernel<<<(n_x + 255) / 256, 256, 0, stream>>>(x,  P0,   n_x);
  cvt_f16_kernel<<<(n_w + 255) / 256, 256, 0, stream>>>(Wq, Wq16, n_w);
  cvt_f16_kernel<<<(n_w + 255) / 256, 256, 0, stream>>>(Wo, Wo16, n_w);
  cvt_f16_kernel<<<(n_w + 255) / 256, 256, 0, stream>>>(W1, W116, n_w);
  cvt_f16_kernel<<<(n_w + 255) / 256, 256, 0, stream>>>(W2, W216, n_w);

  const int gemmBlocks = MROWS / 16;      // 512 blocks x 192 threads (6 waves)

  // q = x @ Wq^T  (f16 out)
  gemm_wmma_kernel<false, false, true><<<gemmBlocks, 192, 0, stream>>>(
      P0, Wq16, nullptr, nullptr, P1);
  // qT for PV B-fragments
  transpose_q_kernel<<<(n_x + 255) / 256, 256, 0, stream>>>(P1, P2);
  // flash attention -> att16 (reuses x16 slot)
  attn_kernel<<<(BATCH * NHEADS * (SEQ / 16)) / 8, 256, 0, stream>>>(P1, P2, P0);
  // sa = att @ Wo^T + bo  (f32)
  gemm_wmma_kernel<true, false, false><<<gemmBlocks, 192, 0, stream>>>(
      P0, Wo16, bo, F0, nullptr);
  // h1 = LN(x + sa)  -> f32 (F1) + f16 (P1)
  add_ln_kernel<<<MROWS, 128, 0, stream>>>(x, F0, g1, be1, F1, P1);
  // ff1 = relu(h1 @ W1^T + b1)  (f16)
  gemm_wmma_kernel<true, true, true><<<gemmBlocks, 192, 0, stream>>>(
      P1, W116, b1, nullptr, P2);
  // ff2 = ff1 @ W2^T + b2  (f32, reuses F0)
  gemm_wmma_kernel<true, false, false><<<gemmBlocks, 192, 0, stream>>>(
      P2, W216, b2, F0, nullptr);
  // out = LN(h1 + ff2)
  add_ln_kernel<<<MROWS, 128, 0, stream>>>(F1, F0, g2, be2, (float*)d_out, nullptr);
}
